// HGTLinkPrediction_46145128628317
// MI455X (gfx1250) — matrix operand order repaired and checked by reading, not compiled, not run on hardware
//
#include <hip/hip_runtime.h>
#include <math.h>

#define NU     50000
#define NPOSTN 100000
#define NEDGE  800000
#define NLBL   200000
#define HID    128
#define HEADS  4
#define DHEAD  32

typedef __attribute__((ext_vector_type(16))) _Float16 v16h;
typedef __attribute__((ext_vector_type(8)))  float    v8f;

// ---------------- order-preserving float<->uint encoding for atomic max ----
__device__ inline unsigned enc_ord(float f) {
  unsigned u = __float_as_uint(f);
  return (u & 0x80000000u) ? ~u : (u | 0x80000000u);
}
__device__ inline float dec_ord(unsigned u) {
  return (u & 0x80000000u) ? __uint_as_float(u ^ 0x80000000u) : __uint_as_float(~u);
}

// ---------------- WMMA GEMM: C[M,N] = act(A[M,K](lda) * W[K,N] + bias) -----
// one wave (32 lanes) per 16x16 output tile; f16 inputs, f32 accumulate.
__global__ void __launch_bounds__(32)
gemm_wmma(const float* __restrict__ A, int lda,
          const float* __restrict__ W, const float* __restrict__ bias,
          float* __restrict__ C, int M, int K, int N, int act) {
  const int lane = threadIdx.x;
  const int row0 = blockIdx.x * 16;
  const int col0 = blockIdx.y * 16;
  const int mloc = lane & 15;
  const int lohi = (lane < 16) ? 0 : 1;

  v8f acc = {};
  for (int kk = 0; kk < K; kk += 32) {
    // A tile 16x32 f16: lane holds row mloc, K = {kb..kb+7, 16+kb..16+kb+7}
    v16h a;
    {
      const int kb = lohi * 8;
      const int row = row0 + mloc;
      if (row < M) {
        const float* ap = A + (size_t)row * lda + kk + kb;
        const float4 f0 = *reinterpret_cast<const float4*>(ap);
        const float4 f1 = *reinterpret_cast<const float4*>(ap + 4);
        const float4 f2 = *reinterpret_cast<const float4*>(ap + 16);
        const float4 f3 = *reinterpret_cast<const float4*>(ap + 20);
        a[0]=(_Float16)f0.x;  a[1]=(_Float16)f0.y;  a[2]=(_Float16)f0.z;  a[3]=(_Float16)f0.w;
        a[4]=(_Float16)f1.x;  a[5]=(_Float16)f1.y;  a[6]=(_Float16)f1.z;  a[7]=(_Float16)f1.w;
        a[8]=(_Float16)f2.x;  a[9]=(_Float16)f2.y;  a[10]=(_Float16)f2.z; a[11]=(_Float16)f2.w;
        a[12]=(_Float16)f3.x; a[13]=(_Float16)f3.y; a[14]=(_Float16)f3.z; a[15]=(_Float16)f3.w;
      } else {
#pragma unroll
        for (int j = 0; j < 16; ++j) a[j] = (_Float16)0.0f;
      }
    }
    // B tile 32x16 f16: lane holds col mloc, K = {ko..ko+7, 16+ko..16+ko+7}
    v16h b;
    {
      const int ko = lohi * 8;
      const float* wp = W + (size_t)(kk + ko) * N + col0 + mloc;
#pragma unroll
      for (int j = 0; j < 16; ++j) b[j] = (_Float16)wp[(size_t)(j < 8 ? j : j + 8) * N];
    }
    acc = __builtin_amdgcn_wmma_f32_16x16x32_f16(false, a, false, b, (short)0, acc, false, false);
  }
  const int n = col0 + mloc;
  const float bval = bias ? bias[n] : 0.0f;
#pragma unroll
  for (int r = 0; r < 8; ++r) {
    const int row = row0 + lohi * 8 + r;
    if (row < M) {
      float v = acc[r] + bval;
      if (act == 1) v = fmaxf(v, 0.0f);
      C[(size_t)row * N + n] = v;
    }
  }
}

// ------------- pack a_rel[4,32,32] into block-diagonal W[128,128] ----------
__global__ void pack_blockdiag(const float* __restrict__ rel, float* __restrict__ W) {
  int t = blockIdx.x * blockDim.x + threadIdx.x;
  if (t >= HID * HID) return;
  int r = t >> 7, c = t & 127;
  int hr = r >> 5, hc = c >> 5;
  W[t] = (hr == hc) ? rel[hr * DHEAD * DHEAD + (r & 31) * DHEAD + (c & 31)] : 0.0f;
}

__global__ void fill_zero(float* __restrict__ p, int n) {
  int t = blockIdx.x * blockDim.x + threadIdx.x;
  if (t < n) p[t] = 0.0f;
}

// ------------- edge pass 1 (wave-per-edge): score + segment max ------------
// 32 lanes cooperatively load the 128-float q row and k_t row (coalesced),
// partial dot per lane, 8-lane tree reduce per head via shfl_xor (wave32).
__global__ void edge_pass1(const float* __restrict__ qbase,   // kqv_dst + 128 (row stride 384)
                           const float* __restrict__ kt,      // [nSrc,128]
                           const int* __restrict__ src, const int* __restrict__ dst,
                           const float* __restrict__ prel,
                           float* __restrict__ score, unsigned* __restrict__ mx, int nE) {
  const int lane = threadIdx.x & 31;
  const int e = blockIdx.x * (blockDim.x >> 5) + (threadIdx.x >> 5);
  if (e >= nE) return;
  const int s = src[e], d = dst[e];
  const float4 qv = reinterpret_cast<const float4*>(qbase + (size_t)d * 384)[lane];
  const float4 kv = reinterpret_cast<const float4*>(kt + (size_t)s * HID)[lane];
  float part = qv.x * kv.x + qv.y * kv.y + qv.z * kv.z + qv.w * kv.w;
  part += __shfl_xor(part, 1, 32);
  part += __shfl_xor(part, 2, 32);
  part += __shfl_xor(part, 4, 32);
  const int h = lane >> 3;                      // 8 lanes per head
  if ((lane & 7) == 0) {
    float sc = part * prel[h] * 0.17677669529663687f;  // 1/sqrt(32)
    score[(size_t)e * HEADS + h] = sc;
    atomicMax(&mx[(size_t)d * 4 + h], enc_ord(sc));
  }
}

// ------------- edge pass 2 (wave-per-edge): exp + coalesced scatter-add ----
__global__ void edge_pass2(const float* __restrict__ score,
                           const float* __restrict__ vt,
                           const int* __restrict__ src, const int* __restrict__ dst,
                           const unsigned* __restrict__ mx,
                           float* __restrict__ sumb, float* __restrict__ numb, int nE) {
  const int lane = threadIdx.x & 31;
  const int e = blockIdx.x * (blockDim.x >> 5) + (threadIdx.x >> 5);
  if (e >= nE) return;
  const int s = src[e], d = dst[e];
  float evl = 0.0f;
  if (lane < HEADS) {
    float m = dec_ord(mx[(size_t)d * 4 + lane]);
    if (!(fabsf(m) < 3.0e38f)) m = 0.0f;        // isfinite guard (matches reference)
    evl = expf(score[(size_t)e * HEADS + lane] - m);
    atomicAdd(&sumb[(size_t)d * 4 + lane], evl);
  }
  const float* vp = vt + (size_t)s * HID;
  float* np = numb + (size_t)d * HID;
#pragma unroll
  for (int c = 0; c < HEADS; ++c) {             // element (c*32+lane) -> head c
    float evc = __shfl(evl, c, 32);
    int idx = c * 32 + lane;
    atomicAdd(&np[idx], evc * vp[idx]);
  }
}

// ------------- normalize + gelu (tanh approx, jax default) ----------------
__global__ void finalize_gelu(const float* __restrict__ numb, const float* __restrict__ sumb,
                              float* __restrict__ g, int nNodes) {
  int t = blockIdx.x * blockDim.x + threadIdx.x;
  if (t >= nNodes * HID) return;
  int node = t >> 7;
  int h = (t >> 5) & 3;
  float s = sumb[(size_t)node * 4 + h];
  float x = numb[t] / (s + 1e-16f);
  float x3 = x * x * x;
  g[t] = 0.5f * x * (1.0f + tanhf(0.7978845608028654f * (x + 0.044715f * x3)));
}

// ------------- gated skip: h = sigmoid(skip)*a + (1-sigmoid)*h ------------
__global__ void skip_blend(const float* __restrict__ a, const float* __restrict__ skp,
                           float* __restrict__ h, int n) {
  int t = blockIdx.x * blockDim.x + threadIdx.x;
  if (t >= n) return;
  float s = 1.0f / (1.0f + expf(-skp[0]));
  h[t] = s * a[t] + (1.0f - s) * h[t];
}

// ------------- fused decoder: gather concat -> 384x64 WMMA -> 64x32 -> 1 ---
__global__ void __launch_bounds__(32)
decoder_kernel(const float* __restrict__ hu,
               const int* __restrict__ esrc, const int* __restrict__ edst,
               const float* __restrict__ w1, const float* __restrict__ b1,
               const float* __restrict__ w2, const float* __restrict__ b2,
               const float* __restrict__ w3, const float* __restrict__ b3,
               float* __restrict__ outp, int nl) {
  __shared__ _Float16 cs[16 * 384];
  __shared__ float    h1s[16 * 64];
  const int lane = threadIdx.x;
  const int row0 = blockIdx.x * 16;

  for (int idx = lane; idx < 16 * 384; idx += 32) {
    int r = idx / 384, col = idx - r * 384;
    int row = row0 + r;
    float v = 0.0f;
    if (row < nl) {
      int sN = esrc[row], dN = edst[row];
      if (col < 128)      v = hu[(size_t)sN * HID + col];
      else if (col < 256) v = hu[(size_t)dN * HID + (col - 128)];
      else {
        int c2 = col - 256;
        v = hu[(size_t)sN * HID + c2] * hu[(size_t)dN * HID + c2];
      }
    }
    cs[idx] = (_Float16)v;
  }
  __syncthreads();

  const int mloc = lane & 15;
  const int lohi = (lane < 16) ? 0 : 1;
  for (int nt = 0; nt < 4; ++nt) {
    v8f acc = {};
    for (int kk = 0; kk < 384; kk += 32) {
      v16h a;
      const int kb = lohi * 8;
      const _Float16* ap = &cs[mloc * 384 + kk + kb];
#pragma unroll
      for (int j = 0; j < 8; ++j) { a[j] = ap[j]; a[8 + j] = ap[16 + j]; }
      v16h b;
      const int ko = lohi * 8;
      const float* wp = w1 + (size_t)(kk + ko) * 64 + nt * 16 + mloc;
#pragma unroll
      for (int j = 0; j < 16; ++j) b[j] = (_Float16)wp[(j < 8 ? j : j + 8) * 64];
      acc = __builtin_amdgcn_wmma_f32_16x16x32_f16(false, a, false, b, (short)0, acc, false, false);
    }
    int n = nt * 16 + mloc;
#pragma unroll
    for (int r = 0; r < 8; ++r)
      h1s[(lohi * 8 + r) * 64 + n] = fmaxf(acc[r] + b1[n], 0.0f);
  }
  __syncthreads();

  if (lane < 16) {
    int row = row0 + lane;
    if (row < nl) {
      float h2[32];
      for (int c2 = 0; c2 < 32; ++c2) {
        float acc = b2[c2];
        for (int j = 0; j < 64; ++j) acc += h1s[lane * 64 + j] * w2[j * 32 + c2];
        h2[c2] = fmaxf(acc, 0.0f);
      }
      float o = b3[0];
      for (int c2 = 0; c2 < 32; ++c2) o += h2[c2] * w3[c2];
      outp[row] = o;
    }
  }
}

// ===========================================================================
extern "C" void kernel_launch(void* const* d_in, const int* in_sizes, int n_in,
                              void* d_out, int out_size, void* d_ws, size_t ws_size,
                              hipStream_t stream) {
  (void)n_in; (void)out_size; (void)ws_size;
  // ---------------- input mapping (robust to flattening order) ------------
  const float *x_user, *x_post;
  const int *efs, *efd, *ews, *ewd, *ers, *erd, *els, *eld;
  const float *lin_w[2], *lin_b[2];   // [0]=user [1]=post
  struct LayerP {
    const float *kqv_w[2], *kqv_b[2];
    const float *a_rel[3], *m_rel[3], *p_rel[3];  // 0=follows 1=writes 2=rev_writes
    const float *out_w[2], *out_b[2], *skip[2];
  } LP[3];
  const float *dw1, *db1, *dw2, *db2, *dw3, *db3;

  int pbase;
  bool outer_insertion = (in_sizes[0] == NU * HID);
  if (outer_insertion) {
    x_user = (const float*)d_in[0];
    x_post = (const float*)d_in[1];
    pbase = 2;
    efs = (const int*)d_in[69]; efd = (const int*)d_in[70];
    ews = (const int*)d_in[71]; ewd = (const int*)d_in[72];
    ers = (const int*)d_in[73]; erd = (const int*)d_in[74];
    els = (const int*)d_in[75]; eld = (const int*)d_in[76];
  } else {  // fully-sorted pytree order
    efd = (const int*)d_in[0]; efs = (const int*)d_in[1];
    erd = (const int*)d_in[2]; ers = (const int*)d_in[3];
    ewd = (const int*)d_in[4]; ews = (const int*)d_in[5];
    eld = (const int*)d_in[6]; els = (const int*)d_in[7];
    pbase = 8;
    x_post = (const float*)d_in[75];
    x_user = (const float*)d_in[76];
  }
  auto P = [&](int off) { return (const float*)d_in[pbase + off]; };
  bool params_sorted = (in_sizes[pbase] == HEADS * DHEAD * DHEAD);  // a_rel first
  if (params_sorted) {
    for (int l = 0; l < 3; ++l) {
      int b = l * 19;
      LP[l].a_rel[0] = P(b + 0); LP[l].a_rel[2] = P(b + 1); LP[l].a_rel[1] = P(b + 2);
      LP[l].kqv_b[1] = P(b + 3); LP[l].kqv_w[1] = P(b + 4);
      LP[l].kqv_b[0] = P(b + 5); LP[l].kqv_w[0] = P(b + 6);
      LP[l].m_rel[0] = P(b + 7); LP[l].m_rel[2] = P(b + 8); LP[l].m_rel[1] = P(b + 9);
      LP[l].out_b[1] = P(b + 10); LP[l].out_w[1] = P(b + 11);
      LP[l].out_b[0] = P(b + 12); LP[l].out_w[0] = P(b + 13);
      LP[l].p_rel[0] = P(b + 14); LP[l].p_rel[2] = P(b + 15); LP[l].p_rel[1] = P(b + 16);
      LP[l].skip[1] = P(b + 17); LP[l].skip[0] = P(b + 18);
    }
    db1 = P(57); db2 = P(58); db3 = P(59); dw1 = P(60); dw2 = P(61); dw3 = P(62);
    lin_b[1] = P(63); lin_w[1] = P(64); lin_b[0] = P(65); lin_w[0] = P(66);
  } else {  // params insertion order: lin, convs, dec
    lin_w[0] = P(0); lin_b[0] = P(1); lin_w[1] = P(2); lin_b[1] = P(3);
    for (int l = 0; l < 3; ++l) {
      int b = 4 + l * 19;
      LP[l].kqv_w[0] = P(b + 0); LP[l].kqv_b[0] = P(b + 1);
      LP[l].kqv_w[1] = P(b + 2); LP[l].kqv_b[1] = P(b + 3);
      LP[l].a_rel[0] = P(b + 4); LP[l].a_rel[1] = P(b + 5); LP[l].a_rel[2] = P(b + 6);
      LP[l].m_rel[0] = P(b + 7); LP[l].m_rel[1] = P(b + 8); LP[l].m_rel[2] = P(b + 9);
      LP[l].p_rel[0] = P(b + 10); LP[l].p_rel[1] = P(b + 11); LP[l].p_rel[2] = P(b + 12);
      LP[l].out_w[0] = P(b + 13); LP[l].out_b[0] = P(b + 14);
      LP[l].out_w[1] = P(b + 15); LP[l].out_b[1] = P(b + 16);
      LP[l].skip[0] = P(b + 17); LP[l].skip[1] = P(b + 18);
    }
    dw1 = P(61); db1 = P(62); dw2 = P(63); db2 = P(64); dw3 = P(65); db3 = P(66);
  }

  // ---------------- workspace carve ---------------------------------------
  float* ws = (float*)d_ws;
  size_t o = 0;
  auto AL = [&](size_t n) { float* p = ws + o; o += n; return p; };
  float* h_u   = AL((size_t)NU * HID);
  float* h_p   = AL((size_t)NPOSTN * HID);
  float* kqv_u = AL((size_t)NU * 3 * HID);
  float* kqv_p = AL((size_t)NPOSTN * 3 * HID);
  float* kt_f  = AL((size_t)NU * HID);
  float* vt_f  = AL((size_t)NU * HID);
  float* kt_w  = AL((size_t)NU * HID);
  float* vt_w  = AL((size_t)NU * HID);
  float* kt_r  = AL((size_t)NPOSTN * HID);
  float* vt_r  = AL((size_t)NPOSTN * HID);
  float* bd[6];
  for (int i = 0; i < 6; ++i) bd[i] = AL((size_t)HID * HID);
  float* accu_u = AL((size_t)NU * (4 + 4 + HID));
  float* accu_p = AL((size_t)NPOSTN * (4 + 4 + HID));
  unsigned* max_u = (unsigned*)accu_u;
  float* sum_u = accu_u + (size_t)NU * 4;
  float* num_u = accu_u + (size_t)NU * 8;
  unsigned* max_p = (unsigned*)accu_p;
  float* sum_p = accu_p + (size_t)NPOSTN * 4;
  float* num_p = accu_p + (size_t)NPOSTN * 8;
  float* sc0 = AL((size_t)NEDGE * 4);
  float* sc1 = AL((size_t)NEDGE * 4);
  float* sc2 = AL((size_t)NEDGE * 4);
  float* g_u = kt_f; float* a_u = vt_f;   // reuse after attention passes
  float* g_p = kt_r; float* a_p = vt_r;

  auto gemm = [&](const float* A, int lda, const float* Wm, const float* bias, float* C,
                  int M, int K, int N, int act) {
    dim3 g((M + 15) / 16, N / 16);
    gemm_wmma<<<g, 32, 0, stream>>>(A, lda, Wm, bias, C, M, K, N, act);
  };

  // ---------------- input projections (relu) ------------------------------
  gemm(x_user, HID, lin_w[0], lin_b[0], h_u, NU, HID, HID, 1);
  gemm(x_post, HID, lin_w[1], lin_b[1], h_p, NPOSTN, HID, HID, 1);

  // wave-per-edge: 256 threads = 8 waves per block
  dim3 eg((NEDGE + 7) / 8);
  for (int l = 0; l < 3; ++l) {
    const LayerP& lp = LP[l];
    gemm(h_u, HID, lp.kqv_w[0], lp.kqv_b[0], kqv_u, NU, HID, 3 * HID, 0);
    gemm(h_p, HID, lp.kqv_w[1], lp.kqv_b[1], kqv_p, NPOSTN, HID, 3 * HID, 0);
    pack_blockdiag<<<64, 256, 0, stream>>>(lp.a_rel[0], bd[0]);
    pack_blockdiag<<<64, 256, 0, stream>>>(lp.m_rel[0], bd[1]);
    pack_blockdiag<<<64, 256, 0, stream>>>(lp.a_rel[1], bd[2]);
    pack_blockdiag<<<64, 256, 0, stream>>>(lp.m_rel[1], bd[3]);
    pack_blockdiag<<<64, 256, 0, stream>>>(lp.a_rel[2], bd[4]);
    pack_blockdiag<<<64, 256, 0, stream>>>(lp.m_rel[2], bd[5]);
    // relation transforms (block-diagonal GEMMs): k cols [0,128), v cols [256,384)
    gemm(kqv_u + 0,       3 * HID, bd[0], nullptr, kt_f, NU, HID, HID, 0);
    gemm(kqv_u + 2 * HID, 3 * HID, bd[1], nullptr, vt_f, NU, HID, HID, 0);
    gemm(kqv_u + 0,       3 * HID, bd[2], nullptr, kt_w, NU, HID, HID, 0);
    gemm(kqv_u + 2 * HID, 3 * HID, bd[3], nullptr, vt_w, NU, HID, HID, 0);
    gemm(kqv_p + 0,       3 * HID, bd[4], nullptr, kt_r, NPOSTN, HID, HID, 0);
    gemm(kqv_p + 2 * HID, 3 * HID, bd[5], nullptr, vt_r, NPOSTN, HID, HID, 0);
    // zero max/sum/num accumulators
    int nu_acc = NU * 136, np_acc = NPOSTN * 136;
    fill_zero<<<(nu_acc + 255) / 256, 256, 0, stream>>>(accu_u, nu_acc);
    fill_zero<<<(np_acc + 255) / 256, 256, 0, stream>>>(accu_p, np_acc);
    // attention pass 1 (scores + segment max)
    edge_pass1<<<eg, 256, 0, stream>>>(kqv_u + HID, kt_f, efs, efd, lp.p_rel[0], sc0, max_u, NEDGE);
    edge_pass1<<<eg, 256, 0, stream>>>(kqv_p + HID, kt_w, ews, ewd, lp.p_rel[1], sc1, max_p, NEDGE);
    edge_pass1<<<eg, 256, 0, stream>>>(kqv_u + HID, kt_r, ers, erd, lp.p_rel[2], sc2, max_u, NEDGE);
    // attention pass 2 (exp + scatter numerator/denominator)
    edge_pass2<<<eg, 256, 0, stream>>>(sc0, vt_f, efs, efd, max_u, sum_u, num_u, NEDGE);
    edge_pass2<<<eg, 256, 0, stream>>>(sc1, vt_w, ews, ewd, max_p, sum_p, num_p, NEDGE);
    edge_pass2<<<eg, 256, 0, stream>>>(sc2, vt_r, ers, erd, max_u, sum_u, num_u, NEDGE);
    // normalize + gelu, output projection, gated skip
    finalize_gelu<<<((NU * HID) + 255) / 256, 256, 0, stream>>>(num_u, sum_u, g_u, NU);
    finalize_gelu<<<((NPOSTN * HID) + 255) / 256, 256, 0, stream>>>(num_p, sum_p, g_p, NPOSTN);
    gemm(g_u, HID, lp.out_w[0], lp.out_b[0], a_u, NU, HID, HID, 0);
    gemm(g_p, HID, lp.out_w[1], lp.out_b[1], a_p, NPOSTN, HID, HID, 0);
    skip_blend<<<((NU * HID) + 255) / 256, 256, 0, stream>>>(a_u, lp.skip[0], h_u, NU * HID);
    skip_blend<<<((NPOSTN * HID) + 255) / 256, 256, 0, stream>>>(a_p, lp.skip[1], h_p, NPOSTN * HID);
  }

  // ---------------- fused link decoder ------------------------------------
  decoder_kernel<<<(NLBL + 15) / 16, 32, 0, stream>>>(
      h_u, els, eld, dw1, db1, dw2, db2, dw3, db3, (float*)d_out, NLBL);
}
